// NonlinearMixture_23510650978887
// MI455X (gfx1250) — matrix operand order repaired and checked by reading, not compiled, not run on hardware
//
#include <hip/hip_runtime.h>
#include <hip/hip_bf16.h>
#include <math.h>

#define Bt 16384
#define Dd 1024
#define Oo 1024
#define Ee 4
#define Kk 32

typedef __bf16 bf16;
typedef bf16 v16bf __attribute__((ext_vector_type(16)));
typedef bf16 v8bf  __attribute__((ext_vector_type(8)));
typedef float v8f  __attribute__((ext_vector_type(8)));

__device__ __forceinline__ v8bf cvt8(float4 a, float4 b) {
    v8bf v;
    v[0] = (bf16)a.x; v[1] = (bf16)a.y; v[2] = (bf16)a.z; v[3] = (bf16)a.w;
    v[4] = (bf16)b.x; v[5] = (bf16)b.y; v[6] = (bf16)b.z; v[7] = (bf16)b.w;
    return v;
}

// ---- CDNA5 async global->LDS (ASYNCcnt-tracked, bypasses VGPRs) ----
__device__ __forceinline__ void async_copy16(void* lds, const void* gptr) {
    // low 32 bits of a generic pointer into LDS == LDS byte offset
    unsigned l = (unsigned)(uintptr_t)lds;
    asm volatile("global_load_async_to_lds_b128 %0, %1, off"
                 :: "v"(l), "v"(gptr) : "memory");
}

#if defined(__has_builtin)
#if __has_builtin(__builtin_amdgcn_s_wait_asynccnt)
#define WAIT_ASYNC(n) __builtin_amdgcn_s_wait_asynccnt(n)
#endif
#endif
#ifndef WAIT_ASYNC
#define WAIT_ASYNC(n) asm volatile("s_wait_asynccnt %0" :: "n"(n) : "memory")
#endif

// ---------------------------------------------------------------- init
__global__ void init_kernel(int* counts, float* zpad) {
    if (threadIdx.x < Ee) counts[threadIdx.x] = 0;
    if (threadIdx.x < 16) zpad[threadIdx.x] = 0.f;
}

// ---------------------------------------------------------------- router
// One wave32 per token. lane k accumulates column-sum over the 32
// non-overlapping windows; logits[e] = sum_k s[k]*w_router[e][k].
__global__ __launch_bounds__(256) void router_kernel(
    const float* __restrict__ x, const float* __restrict__ wr,
    float* __restrict__ sel_out,          // d_out one-hot dispatch [B,E]
    float* __restrict__ gate_ws,          // [B]
    float* __restrict__ sel_ws,           // [B,E] softmax values
    int* __restrict__ counts,             // [E]
    int* __restrict__ lists)              // [E,B]
{
    const int lane = threadIdx.x & 31;
    const int wave = threadIdx.x >> 5;
    const int b = blockIdx.x * 8 + wave;
    if (b >= Bt) return;

    const float* xr = x + (size_t)b * Dd;
    float s = 0.f;
    #pragma unroll
    for (int p = 0; p < Dd / Kk; ++p) s += xr[p * Kk + lane];

    float l0 = s * wr[0 * Kk + lane];
    float l1 = s * wr[1 * Kk + lane];
    float l2 = s * wr[2 * Kk + lane];
    float l3 = s * wr[3 * Kk + lane];
    #pragma unroll
    for (int off = 16; off > 0; off >>= 1) {
        l0 += __shfl_xor(l0, off, 32);
        l1 += __shfl_xor(l1, off, 32);
        l2 += __shfl_xor(l2, off, 32);
        l3 += __shfl_xor(l3, off, 32);
    }

    if (lane == 0) {
        float l[Ee] = {l0, l1, l2, l3};
        int idx = 0; float mx = l[0];
        #pragma unroll
        for (int e = 1; e < Ee; ++e) if (l[e] > mx) { mx = l[e]; idx = e; }
        float se[Ee]; float sum = 0.f;
        #pragma unroll
        for (int e = 0; e < Ee; ++e) { se[e] = __expf(l[e] - mx); sum += se[e]; }
        const float inv = 1.f / sum;
        #pragma unroll
        for (int e = 0; e < Ee; ++e) {
            sel_ws[(size_t)b * Ee + e]  = se[e] * inv;
            sel_out[(size_t)b * Ee + e] = (e == idx) ? 1.f : 0.f;
        }
        gate_ws[b] = se[idx] * inv;
        int pos = atomicAdd(&counts[idx], 1);
        lists[idx * Bt + pos] = b;
    }
}

// ---------------------------------------------------------------- loss
// Fixed-order reduction (deterministic, no float atomics).
__global__ __launch_bounds__(256) void loss_kernel(
    const float* __restrict__ sel_ws, const int* __restrict__ counts,
    float* __restrict__ loss_out)
{
    __shared__ float red[256][Ee];
    float a[Ee] = {0.f, 0.f, 0.f, 0.f};
    const float4* s4 = (const float4*)sel_ws;
    for (int i = threadIdx.x; i < Bt; i += 256) {
        float4 v = s4[i];
        a[0] += v.x; a[1] += v.y; a[2] += v.z; a[3] += v.w;
    }
    #pragma unroll
    for (int e = 0; e < Ee; ++e) red[threadIdx.x][e] = a[e];
    __syncthreads();
    for (int o = 128; o > 0; o >>= 1) {
        if (threadIdx.x < o) {
            #pragma unroll
            for (int e = 0; e < Ee; ++e)
                red[threadIdx.x][e] += red[threadIdx.x + o][e];
        }
        __syncthreads();
    }
    if (threadIdx.x == 0) {
        float acc = 0.f;
        #pragma unroll
        for (int e = 0; e < Ee; ++e) acc += red[0][e] * (float)counts[e];
        *loss_out = acc * (float)Ee / ((float)Bt * (float)Bt);
    }
}

// ---------------------------------------------------------------- expert GEMM
// Compacted per-expert GEMM via v_wmma_f32_16x16x32_bf16 with double-buffered
// async global->LDS staging (fp32 tiles in LDS; cvt to bf16 at frag build,
// co-executing with the XDL WMMA pipe).
// Block tile: 64 tokens x 128 outputs, K-step 32. 8 waves, each owns a 16x64
// strip (4 WMMA accumulators).
#define A_STRIDE 36   // fp32 elements per LDS row (padded: conflict-free)
#define B_STRIDE 36

__global__ __launch_bounds__(256) void expert_gemm_kernel(
    const float* __restrict__ x, const float* __restrict__ Wexp,
    const float* __restrict__ bexp, const float* __restrict__ gate_ws,
    const int* __restrict__ counts, const int* __restrict__ lists,
    const float* __restrict__ zpad, float* __restrict__ out)
{
    const int e      = blockIdx.z;
    const int cnt    = counts[e];
    const int tstart = blockIdx.y * 64;
    if (tstart >= cnt) return;            // uniform per block
    const int nbase = blockIdx.x * 128;

    __shared__ __align__(16) float Asf[2][64][A_STRIDE];
    __shared__ __align__(16) float Bsf[2][128][B_STRIDE];
    __shared__ int toks[64];

    const int tid = threadIdx.x;
    if (tid < 64) {
        int r = tstart + tid;
        toks[tid] = (r < cnt) ? lists[e * Bt + r] : -1;
    }
    __syncthreads();

    const float* We = Wexp + (size_t)e * Oo * Dd;

    // per-thread staging coordinates (6 x 16B async chunks per thread/K-step)
    const int arow = tid >> 2;            // A: 64 rows x 8 chunks (2/thread)
    const int aq   = (tid & 3) * 2;
    const int atok = toks[arow];
    const float* agbase = (atok >= 0) ? (x + (size_t)atok * Dd + aq * 4) : zpad;
    const int astep = (atok >= 0) ? 32 : 0;   // invalid rows keep reading zpad

    const int lane = tid & 31;
    const int wave = tid >> 5;
    const int m0 = (wave & 3) * 16;   // token sub-tile
    const int nh = wave >> 2;         // 0/1 : which 64-wide output half
    const int lh = lane & 15;
    const int lg = lane >> 4;

    v8f acc[4];
    #pragma unroll
    for (int s = 0; s < 4; ++s)
        #pragma unroll
        for (int i = 0; i < 8; ++i) acc[s][i] = 0.f;

    const int NK = Dd / 32;
    int cur = 0;

    // ---- issue async stage for a K-step into buffer `buf` ----
    auto stage = [&](int buf, int kb) {
        const float* ga = agbase + (size_t)kb * (astep ? 1 : 0) * 0; // see below
        // A tile (gathered rows; invalid rows source a zeroed pad)
        const float* g0 = (atok >= 0) ? (x + (size_t)atok * Dd + kb + aq * 4) : zpad;
        const float* g1 = (atok >= 0) ? (g0 + 4) : zpad;
        async_copy16(&Asf[buf][arow][aq * 4], g0);
        async_copy16(&Asf[buf][arow][aq * 4 + 4], g1);
        (void)ga;
        // B tile: 128 rows x 8 chunks (4/thread); W rows are contiguous
        #pragma unroll
        for (int j = 0; j < 4; ++j) {
            const int c = tid + 256 * j;
            const int row = c >> 3, q = c & 7;
            async_copy16(&Bsf[buf][row][q * 4],
                         We + (size_t)(nbase + row) * Dd + kb + q * 4);
        }
    };

    stage(0, 0);
    for (int i = 0; i < NK; ++i) {
        if (i + 1 < NK) {
            stage(cur ^ 1, (i + 1) * 32);   // prefetch next K-step
            WAIT_ASYNC(6);                  // oldest 6 (cur buffer) done
        } else {
            WAIT_ASYNC(0);
        }
        __syncthreads();

        // --- fragment build (fp32 LDS -> bf16) + 4 WMMAs ---
        const int Mr = m0 + lh;
        const float4* ap  = (const float4*)&Asf[cur][Mr][lg * 8];
        const float4* ap2 = (const float4*)&Asf[cur][Mr][lg * 8 + 16];
        v8bf alo = cvt8(ap[0], ap[1]);
        v8bf ahi = cvt8(ap2[0], ap2[1]);
        v16bf a;
        #pragma unroll
        for (int k = 0; k < 8; ++k) { a[k] = alo[k]; a[8 + k] = ahi[k]; }

        #pragma unroll
        for (int s = 0; s < 4; ++s) {
            const int Nr = nh * 64 + s * 16 + lh;
            const float4* bp = (const float4*)&Bsf[cur][Nr][lg * 8];
            v8bf blo = cvt8(bp[0], bp[1]);
            v8bf bhi = cvt8(bp[4], bp[5]);
            v16bf bb;
            #pragma unroll
            for (int k = 0; k < 8; ++k) { bb[k] = blo[k]; bb[8 + k] = bhi[k]; }
            acc[s] = __builtin_amdgcn_wmma_f32_16x16x32_bf16(
                false, a, false, bb, (short)0, acc[s], false, false);
        }
        __syncthreads();        // all reads of `cur` done before next overwrite
        cur ^= 1;
    }

    // --- epilogue: bias, cube, gate, scatter ---
    #pragma unroll
    for (int s = 0; s < 4; ++s) {
        const int gn = nbase + nh * 64 + s * 16 + lh;
        const float bias = bexp[e * Oo + gn];
        #pragma unroll
        for (int r = 0; r < 8; ++r) {
            const int row = m0 + r + lg * 8;
            const int tok = toks[row];
            if (tok >= 0) {
                float h = acc[s][r] + bias;
                out[(size_t)tok * Oo + gn] = gate_ws[tok] * h * h * h;
            }
        }
    }
}

// ---------------------------------------------------------------- row softmax
__global__ __launch_bounds__(256) void softmax_kernel(float* __restrict__ out)
{
    __shared__ float red[256];
    float* row = out + (size_t)blockIdx.x * Oo;
    const int t = threadIdx.x;
    float v0 = row[t], v1 = row[t + 256], v2 = row[t + 512], v3 = row[t + 768];
    float m = fmaxf(fmaxf(v0, v1), fmaxf(v2, v3));
    red[t] = m; __syncthreads();
    for (int o = 128; o > 0; o >>= 1) {
        if (t < o) red[t] = fmaxf(red[t], red[t + o]);
        __syncthreads();
    }
    m = red[0]; __syncthreads();
    v0 = __expf(v0 - m); v1 = __expf(v1 - m);
    v2 = __expf(v2 - m); v3 = __expf(v3 - m);
    red[t] = v0 + v1 + v2 + v3; __syncthreads();
    for (int o = 128; o > 0; o >>= 1) {
        if (t < o) red[t] += red[t + o];
        __syncthreads();
    }
    const float inv = 1.f / red[0];
    row[t] = v0 * inv; row[t + 256] = v1 * inv;
    row[t + 512] = v2 * inv; row[t + 768] = v3 * inv;
}

// ---------------------------------------------------------------- launch
extern "C" void kernel_launch(void* const* d_in, const int* in_sizes, int n_in,
                              void* d_out, int out_size, void* d_ws, size_t ws_size,
                              hipStream_t stream) {
    const float* x  = (const float*)d_in[0];   // [B,D]
    const float* wr = (const float*)d_in[1];   // [E,K]
    const float* We = (const float*)d_in[2];   // [E,O,D]
    const float* be = (const float*)d_in[3];   // [E,O]

    float* out      = (float*)d_out;                    // [B,O]
    float* sel_out  = out + (size_t)Bt * Oo;            // [B,E]
    float* loss_out = sel_out + (size_t)Bt * Ee;        // [1]

    // workspace layout
    float* gate_ws = (float*)d_ws;                      // [B]
    float* sel_ws  = gate_ws + Bt;                      // [B,E]
    int*   counts  = (int*)(sel_ws + (size_t)Bt * Ee);  // [E]
    int*   lists   = counts + 8;                        // [E,B]
    float* zpad    = (float*)(lists + (size_t)Ee * Bt); // [16] zeros

    init_kernel<<<1, 64, 0, stream>>>(counts, zpad);
    router_kernel<<<Bt / 8, 256, 0, stream>>>(x, wr, sel_out, gate_ws, sel_ws,
                                              counts, lists);
    loss_kernel<<<1, 256, 0, stream>>>(sel_ws, counts, loss_out);
    expert_gemm_kernel<<<dim3(Oo / 128, Bt / 64, Ee), 256, 0, stream>>>(
        x, We, be, gate_ws, counts, lists, zpad, out);
    softmax_kernel<<<Bt, 256, 0, stream>>>(out);
}
